// AttentionBlock_3908420239929
// MI455X (gfx1250) — compile-verified, matrix operand order checked by reading, not run on hardware
//
#include <hip/hip_runtime.h>
#include <hip/hip_bf16.h>

#define INPUT_DIM 256
#define HID 128
#define NH 8
#define SEQ 1024
#define BATCHN 8
#define CDIM 1024  /* NH*HID */
#define NCHUNK (SEQ / 32)

typedef __attribute__((ext_vector_type(16))) __bf16 v16bf;
typedef __attribute__((ext_vector_type(8)))  float  v8f;
typedef __attribute__((ext_vector_type(4)))  int    v4i;
typedef __attribute__((address_space(1))) v4i as1_v4i;
typedef __attribute__((address_space(3))) v4i as3_v4i;

#if defined(__HIP_DEVICE_COMPILE__) && __has_builtin(__builtin_amdgcn_global_load_async_to_lds_b128)
#define HAVE_ASYNC 1
#else
#define HAVE_ASYNC 0
#endif

#if HAVE_ASYNC
#if __has_builtin(__builtin_amdgcn_s_wait_asynccnt)
#define WAIT_ASYNC(n) __builtin_amdgcn_s_wait_asynccnt(n)
#else
#define WAIT_ASYNC(n) asm volatile("s_wait_asynccnt " #n ::: "memory")
#endif
#else
#define WAIT_ASYNC(n) ((void)0)
#endif

__device__ __forceinline__ __bf16 f2bf(float f) {
    union { float f; unsigned int u; } a; a.f = f;
    unsigned int u = a.u;
    u += 0x7fffu + ((u >> 16) & 1u);           // round-to-nearest-even
    union { unsigned short s; __bf16 b; } r; r.s = (unsigned short)(u >> 16);
    return r.b;
}

union FragU { v16bf v; uint4 q[2]; };

// 16x32 bf16 WMMA operand fragment (A layout; B uses the same layout with
// lane = output column, reading rows of the transposed matrix).
// Lanes 0-15: K=0..7 (v0..3), K=16..23 (v4..7) of row `lane`;
// lanes 16-31: K=8..15 and K=24..31 of row `lane-16`.
__device__ __forceinline__ v16bf load_frag(const __bf16* base, int stride) {
    const int lane = threadIdx.x & 31;
    const int r    = lane & 15;
    const int sel  = lane >> 4;
    const __bf16* p = base + (size_t)r * stride + sel * 8;
    FragU f;
    f.q[0] = *(const uint4*)(p);
    f.q[1] = *(const uint4*)(p + 16);
    return f.v;
}

__device__ __forceinline__ v8f wmma_bf16(v16bf a, v16bf b, v8f c) {
    return __builtin_amdgcn_wmma_f32_16x16x32_bf16(
        false, a, false, b, (short)0, c, false, false);
}

// 16-byte global -> LDS copy; async (ASYNCcnt) when available.
__device__ __forceinline__ void async_copy16(__bf16* ldst, const __bf16* gsrc) {
#if HAVE_ASYNC
    __builtin_amdgcn_global_load_async_to_lds_b128(
        (as1_v4i*)(uintptr_t)gsrc,
        (as3_v4i*)(uintptr_t)ldst, 0, 0);
#else
    *(uint4*)ldst = *(const uint4*)gsrc;
#endif
}

// ---------------------------------------------------------------- converters
__global__ void cvt_f32_bf16(const float* __restrict__ in, __bf16* __restrict__ out, int n) {
    int i = blockIdx.x * blockDim.x + threadIdx.x;
    if (i < n) out[i] = f2bf(in[i]);
}

// in [K][Ncols] (row major) -> out [Ncols][K] bf16
__global__ void tpose_cvt(const float* __restrict__ in, __bf16* __restrict__ out,
                          int K, int Ncols) {
    int idx = blockIdx.x * blockDim.x + threadIdx.x;
    if (idx < K * Ncols) {
        int k = idx / Ncols, c = idx % Ncols;
        out[(size_t)c * K + k] = f2bf(in[idx]);
    }
}

// ------------------------------------------------------------ fused QKV GEMM
__global__ __launch_bounds__(128)
void qkv_gemm(const __bf16* __restrict__ Xb,
              const __bf16* __restrict__ Wqt, const __bf16* __restrict__ Wkt,
              const __bf16* __restrict__ Wvt,
              const float* __restrict__ bq, const float* __restrict__ bk,
              const float* __restrict__ bv,
              __bf16* __restrict__ Qh, __bf16* __restrict__ Kh, __bf16* __restrict__ Vt)
{
    const int wave = threadIdx.x >> 5;
    const int lane = threadIdx.x & 31;
    const int half = lane >> 4;
    const int lc   = lane & 15;
    const int z    = blockIdx.z;
    const __bf16* Wt   = (z == 0) ? Wqt : (z == 1) ? Wkt : Wvt;
    const float*  bias = (z == 0) ? bq  : (z == 1) ? bk  : bv;
    const int r0 = blockIdx.y * 64 + wave * 16;
    const int c0 = blockIdx.x * 64;

    v8f acc[4] = {};
    for (int kc = 0; kc < 8; ++kc) {
        const int k0 = kc * 32;
        v16bf af = load_frag(Xb + (size_t)r0 * INPUT_DIM + k0, INPUT_DIM);
#pragma unroll
        for (int t = 0; t < 4; ++t) {
            v16bf bf = load_frag(Wt + (size_t)(c0 + 16 * t) * INPUT_DIM + k0, INPUT_DIM);
            acc[t] = wmma_bf16(af, bf, acc[t]);
        }
    }
#pragma unroll
    for (int t = 0; t < 4; ++t) {
#pragma unroll
        for (int i = 0; i < 8; ++i) {
            const int row = r0 + i + 8 * half;     // (b, n)
            const int col = c0 + 16 * t + lc;      // (h, d)
            const __bf16 vbf = f2bf(acc[t][i] + bias[col]);
            const int b = row >> 10, n = row & 1023;
            const int h = col >> 7,  d = col & 127;
            if (z == 2)
                Vt[((size_t)(b * NH + h) * HID + d) * SEQ + n] = vbf;   // transposed V
            else {
                __bf16* O = (z == 0) ? Qh : Kh;
                O[((size_t)(b * NH + h) * SEQ + n) * HID + d] = vbf;
            }
        }
    }
}

// ------------------------------------------------------- flash attention core
// Block = 4 waves x 16 rows of one (b,h); K/V chunks double-buffered in LDS
// via async global->LDS copies shared by all 4 waves.
__global__ __launch_bounds__(128)
void flash_attn(const __bf16* __restrict__ Qh, const __bf16* __restrict__ Kh,
                const __bf16* __restrict__ Vt, const float* __restrict__ Bbias,
                __bf16* __restrict__ Ob)
{
    __shared__ __bf16 Kbuf[2][32 * 128];   // K rows m0..m0+31, row-major stride 128
    __shared__ __bf16 Vbuf[2][128 * 32];   // V^T rows d=0..127, cols m0..m0+31, stride 32
    __shared__ __bf16 ldsP[4 * 16 * 32];   // per-wave 16x32 P tile

    const int tid  = threadIdx.x;
    const int wave = tid >> 5;
    const int lane = tid & 31;
    const int half = lane >> 4;
    const int lc   = lane & 15;
    const int bh   = blockIdx.y;
    const int n0   = blockIdx.x * 64 + wave * 16;
    const int b    = bh >> 3;
    const int h    = bh & 7;

    const __bf16* Qbase = Qh + ((size_t)bh * SEQ + n0) * HID;
    const __bf16* Kbase = Kh + (size_t)bh * SEQ * HID;
    const __bf16* Vbase = Vt + (size_t)bh * HID * SEQ;

    v16bf qf[4];
#pragma unroll
    for (int kc = 0; kc < 4; ++kc) qf[kc] = load_frag(Qbase + kc * 32, HID);

    v8f oacc[8] = {};
    float mi[8], li[8];
#pragma unroll
    for (int i = 0; i < 8; ++i) { mi[i] = -__builtin_inff(); li[i] = 0.f; }

    __bf16* P = ldsP + wave * 16 * 32;
    const float scale = 0.088388347648318447f;    // 1/sqrt(128)

    // cooperative stage of one 32-column chunk (8 x b128 per thread)
    auto stage = [&](int buf, int m0) {
        const __bf16* gk = Kbase + (size_t)m0 * HID;   // 8192B contiguous
#pragma unroll
        for (int q = 0; q < 4; ++q) {
            const int o = (q * 128 + tid) * 8;
            async_copy16(&Kbuf[buf][o], gk + o);
        }
        const __bf16* gv = Vbase + (size_t)tid * SEQ + m0;  // one V^T row / thread
#pragma unroll
        for (int s = 0; s < 4; ++s)
            async_copy16(&Vbuf[buf][tid * 32 + s * 8], gv + s * 8);
    };

    stage(0, 0);
    for (int mc = 0; mc < NCHUNK; ++mc) {
        const int cur = mc & 1;
        const int m0  = mc * 32;
        if (mc + 1 < NCHUNK) {
            stage(cur ^ 1, m0 + 32);   // prefetch next chunk
            WAIT_ASYNC(8);             // 8 newest outstanding -> current chunk done
        } else {
            WAIT_ASYNC(0);
        }
        __syncthreads();               // all waves' async data visible

        v8f s0 = {}, s1 = {};
#pragma unroll
        for (int kc = 0; kc < 4; ++kc)
            s0 = wmma_bf16(qf[kc], load_frag(&Kbuf[cur][0 * 128 + kc * 32], 128), s0);
#pragma unroll
        for (int kc = 0; kc < 4; ++kc)
            s1 = wmma_bf16(qf[kc], load_frag(&Kbuf[cur][16 * 128 + kc * 32], 128), s1);

#pragma unroll
        for (int i = 0; i < 8; ++i) {
            const int row = n0 + i + 8 * half;
            const float x0 = s0[i] * scale + Bbias[(size_t)row * SEQ + m0 + lc];
            const float x1 = s1[i] * scale + Bbias[(size_t)row * SEQ + m0 + 16 + lc];
            float rm = fmaxf(x0, x1);
            rm = fmaxf(rm, __shfl_xor(rm, 1, 32));
            rm = fmaxf(rm, __shfl_xor(rm, 2, 32));
            rm = fmaxf(rm, __shfl_xor(rm, 4, 32));
            rm = fmaxf(rm, __shfl_xor(rm, 8, 32));
            const float mnew = fmaxf(mi[i], rm);
            const float corr = __expf(mi[i] - mnew);
            const float p0 = __expf(x0 - mnew);
            const float p1 = __expf(x1 - mnew);
            float rs = p0 + p1;
            rs += __shfl_xor(rs, 1, 32);
            rs += __shfl_xor(rs, 2, 32);
            rs += __shfl_xor(rs, 4, 32);
            rs += __shfl_xor(rs, 8, 32);
            li[i] = li[i] * corr + rs;
            mi[i] = mnew;
#pragma unroll
            for (int j = 0; j < 8; ++j) oacc[j][i] *= corr;
            const int prow = i + 8 * half;
            P[prow * 32 + lc]      = f2bf(p0);
            P[prow * 32 + 16 + lc] = f2bf(p1);
        }
        asm volatile("s_wait_dscnt 0" ::: "memory");   // P visible across lanes
        v16bf pf = load_frag(P, 32);
#pragma unroll
        for (int j = 0; j < 8; ++j) {
            v16bf vf = load_frag(&Vbuf[cur][(j * 16) * 32], 32);
            oacc[j] = wmma_bf16(pf, vf, oacc[j]);
        }
        __syncthreads();               // done consuming before buffer reuse
    }

    // merge heads: Ob[b][n][h*128 + d]
#pragma unroll
    for (int j = 0; j < 8; ++j)
#pragma unroll
        for (int i = 0; i < 8; ++i) {
            const int n = n0 + i + 8 * half;
            Ob[((size_t)b * SEQ + n) * CDIM + h * HID + j * 16 + lc] =
                f2bf(oacc[j][i] / li[i]);
        }
}

// ------------------------------------------------------------ final projection
__global__ __launch_bounds__(128)
void proj_gemm(const __bf16* __restrict__ Ob, const __bf16* __restrict__ Pt,
               const float* __restrict__ pb, float* __restrict__ out)
{
    const int wave = threadIdx.x >> 5;
    const int lane = threadIdx.x & 31;
    const int half = lane >> 4;
    const int lc   = lane & 15;
    const int r0   = blockIdx.x * 64 + wave * 16;

    v8f acc[8] = {};
    for (int kc = 0; kc < 32; ++kc) {
        const int k0 = kc * 32;
        v16bf af = load_frag(Ob + (size_t)r0 * CDIM + k0, CDIM);
#pragma unroll
        for (int j = 0; j < 8; ++j) {
            v16bf bf = load_frag(Pt + (size_t)(j * 16) * CDIM + k0, CDIM);
            acc[j] = wmma_bf16(af, bf, acc[j]);
        }
    }
#pragma unroll
    for (int j = 0; j < 8; ++j)
#pragma unroll
        for (int i = 0; i < 8; ++i) {
            const int row = r0 + i + 8 * half;
            const int col = j * 16 + lc;
            out[(size_t)row * HID + col] = acc[j][i] + pb[col];
        }
}

// ----------------------------------------------------------------- launcher
extern "C" void kernel_launch(void* const* d_in, const int* in_sizes, int n_in,
                              void* d_out, int out_size, void* d_ws, size_t ws_size,
                              hipStream_t stream) {
    const float* x  = (const float*)d_in[0];
    const float* Bb = (const float*)d_in[1];
    const float* wq = (const float*)d_in[2];
    const float* bq = (const float*)d_in[3];
    const float* wk = (const float*)d_in[4];
    const float* bk = (const float*)d_in[5];
    const float* wv = (const float*)d_in[6];
    const float* bv = (const float*)d_in[7];
    const float* pw = (const float*)d_in[8];
    const float* pb = (const float*)d_in[9];
    float* out = (float*)d_out;
    (void)in_sizes; (void)n_in; (void)out_size; (void)ws_size;

    char* ws = (char*)d_ws;
    size_t off = 0;
    auto carve = [&](size_t bytes) -> char* {
        char* p = ws + off;
        off += (bytes + 255) & ~(size_t)255;
        return p;
    };
    __bf16* Xb  = (__bf16*)carve(sizeof(__bf16) * (size_t)BATCHN * SEQ * INPUT_DIM);
    __bf16* Wqt = (__bf16*)carve(sizeof(__bf16) * (size_t)CDIM * INPUT_DIM);
    __bf16* Wkt = (__bf16*)carve(sizeof(__bf16) * (size_t)CDIM * INPUT_DIM);
    __bf16* Wvt = (__bf16*)carve(sizeof(__bf16) * (size_t)CDIM * INPUT_DIM);
    __bf16* Pt  = (__bf16*)carve(sizeof(__bf16) * (size_t)HID * CDIM);
    __bf16* Qh  = (__bf16*)carve(sizeof(__bf16) * (size_t)BATCHN * NH * SEQ * HID);
    __bf16* Kh  = (__bf16*)carve(sizeof(__bf16) * (size_t)BATCHN * NH * SEQ * HID);
    __bf16* Vt  = (__bf16*)carve(sizeof(__bf16) * (size_t)BATCHN * NH * HID * SEQ);
    __bf16* Ob  = (__bf16*)carve(sizeof(__bf16) * (size_t)BATCHN * SEQ * CDIM);

    const int nx = BATCHN * SEQ * INPUT_DIM;
    cvt_f32_bf16<<<(nx + 255) / 256, 256, 0, stream>>>(x, Xb, nx);
    const int nw = INPUT_DIM * CDIM;
    tpose_cvt<<<(nw + 255) / 256, 256, 0, stream>>>(wq, Wqt, INPUT_DIM, CDIM);
    tpose_cvt<<<(nw + 255) / 256, 256, 0, stream>>>(wk, Wkt, INPUT_DIM, CDIM);
    tpose_cvt<<<(nw + 255) / 256, 256, 0, stream>>>(wv, Wvt, INPUT_DIM, CDIM);
    const int np = CDIM * HID;
    tpose_cvt<<<(np + 255) / 256, 256, 0, stream>>>(pw, Pt, CDIM, HID);

    qkv_gemm<<<dim3(CDIM / 64, (BATCHN * SEQ) / 64, 3), 128, 0, stream>>>(
        Xb, Wqt, Wkt, Wvt, bq, bk, bv, Qh, Kh, Vt);

    flash_attn<<<dim3(SEQ / 64, BATCHN * NH), 128, 0, stream>>>(Qh, Kh, Vt, Bb, Ob);

    proj_gemm<<<dim3((BATCHN * SEQ) / 64), 128, 0, stream>>>(Ob, Pt, pb, out);
}
